// MnistCapsuleModel_32959579030395
// MI455X (gfx1250) — compile-verified, matrix operand order checked by reading, not run on hardware
//
#include <hip/hip_runtime.h>
#include <hip/hip_bf16.h>
#include <math.h>

// ---------------------------------------------------------------------------
// CapsNet forward for MI455X (gfx1250, wave32).
// conv2 / primary-caps are implicit GEMMs on v_wmma_f32_16x16x32_f16.
// Block = 256 thr (8 waves): M-tile 256, N-tile 64, K-step 32, LDS dbl-buffer.
// K dimension permuted to (ky,kx,ic) so the im2col fill is shift/mask +
// constant-offset loads. Each wave: 2 A-frags x 4 B-frags -> 8 WMMA / K-step.
// ---------------------------------------------------------------------------

typedef __attribute__((ext_vector_type(16))) _Float16 v16h;
typedef __attribute__((ext_vector_type(8)))  _Float16 v8h;
typedef __attribute__((ext_vector_type(8)))  float    v8f;

// ----------------------------- conv1 + BN + ReLU ---------------------------
__global__ __launch_bounds__(256)
void k_conv1(const float* __restrict__ x, const float* __restrict__ w,
             const float* __restrict__ bias, const float* __restrict__ g,
             const float* __restrict__ be, const float* __restrict__ mu,
             const float* __restrict__ var, _Float16* __restrict__ h1)
{
  int idx = blockIdx.x * 256 + threadIdx.x;              // (b,c,y,x) flat
  if (idx >= 512 * 128 * 784) return;
  int p = idx % 784;
  int t = idx / 784;
  int c = t % 128;
  int b = t / 128;
  int y = p / 28, xx = p % 28;
  const float* xin = x + (size_t)b * 784;                // single input channel
  const float* wc  = w + c * 9;
  float acc = 0.f;
  #pragma unroll
  for (int ky = 0; ky < 3; ++ky) {
    int iy = y + ky - 1;
    if (iy < 0 || iy >= 28) continue;
    #pragma unroll
    for (int kx = 0; kx < 3; ++kx) {
      int ix = xx + kx - 1;
      if (ix < 0 || ix >= 28) continue;
      acc += wc[ky * 3 + kx] * xin[iy * 28 + ix];
    }
  }
  acc += bias[c];
  float sc = g[c] * rsqrtf(var[c] + 1e-5f);
  acc = (acc - mu[c]) * sc + be[c];
  acc = fmaxf(acc, 0.f);
  h1[idx] = (_Float16)acc;
}

// ------------------- weight repack: (oc,ic,kk) -> (oc,kk,ic), f32->f16 -----
__global__ __launch_bounds__(256)
void k_repack_w(const float* __restrict__ in, _Float16* __restrict__ out, int IC)
{
  int idx = blockIdx.x * 256 + threadIdx.x;
  int total = 256 * IC * 81;
  if (idx >= total) return;
  int kk = idx % 81;
  int t  = idx / 81;
  int ic = t % IC;
  int oc = t / IC;
  out[(size_t)oc * IC * 81 + (size_t)kk * IC + ic] = (_Float16)in[idx];
}

// --------------------- implicit-GEMM conv via WMMA -------------------------
// MODE 0: conv2  (IC=128, 28x28 -> 20x20, stride 1, +bias, ReLU, f16 out)
// MODE 1: pcaps  (IC=256, 20x20 ->  6x6, stride 2, +bias, f32 out)
// GEMM K index = kk*IC + ic  with kk = ky*9+kx  (power-of-two IC).
template <int MODE>
__global__ __launch_bounds__(256)
void k_convgemm(const _Float16* __restrict__ Wm, const _Float16* __restrict__ In,
                const float* __restrict__ bias, _Float16* __restrict__ OutH,
                float* __restrict__ OutF)
{
  constexpr int IC   = (MODE == 0) ? 128 : 256;
  constexpr int ICL2 = (MODE == 0) ? 7 : 8;
  constexpr int IH   = (MODE == 0) ? 28 : 20;
  constexpr int OH   = (MODE == 0) ? 20 : 6;
  constexpr int STR  = (MODE == 0) ? 1 : 2;
  constexpr int K    = IC * 81;
  constexpr int NP   = OH * OH;                // spatial positions per image
  constexpr int BST  = 40;                     // LDS [n][k] stride (16B aligned)

  __shared__ __align__(16) _Float16 Bt[2][64 * BST];   // double-buffered B tile

  const int nBase = blockIdx.x * 64;
  const int lane  = threadIdx.x & 31;
  const int wave  = threadIdx.x >> 5;
  const int mRow  = wave * 16 + (lane & 15);           // A-frag row (m-sub 0)
  const int kb    = (lane >> 4) * 8;                   // per-lane k sub-base

  // --- fill assignment: thread owns (n = tid>>2, k-octet = (tid&3)*8) ------
  const int fn  = threadIdx.x >> 2;                    // 0..63
  const int fko = (threadIdx.x & 3) * 8;               // 0,8,16,24
  const int fng = nBase + fn;
  const int fb  = fng / NP, fs = fng - fb * NP;
  const int foy = fs / OH,  fox = fs - foy * OH;
  const _Float16* fIn = In + (size_t)fb * IC * IH * IH
                           + (size_t)(foy * STR) * IH + (fox * STR);

  // gather one k-octet (8 consecutive ic's, same (ky,kx)): constant offsets
  auto gather = [&](int kg) -> v8h {
    int kk  = kg >> ICL2;                 // kernel position ky*9+kx
    int ic0 = kg & (IC - 1);              // octet start channel
    int ky  = kk / 9, kx = kk - ky * 9;
    const _Float16* src = fIn + (size_t)ic0 * IH * IH + ky * IH + kx;
    v8h r;
    #pragma unroll
    for (int q = 0; q < 8; ++q) r[q] = src[(size_t)q * IH * IH];
    return r;
  };

  // prologue: stage k-step 0
  {
    v8h t0 = gather(fko);
    *(v8h*)&Bt[0][fn * BST + fko] = t0;
  }
  __syncthreads();

  v8f acc[2][4] = {};
  int p = 0;
  for (int k0 = 0; k0 < K; k0 += 32) {
    const int kn = k0 + 32;
    v8h nxt;
    if (kn < K) nxt = gather(kn + fko);    // in flight during compute

    // two A fragments (m and m+128), contiguous in K
    const v8h* a0p = (const v8h*)(Wm + (size_t)mRow * K + k0 + kb);
    const v8h* a1p = (const v8h*)(Wm + (size_t)(mRow + 128) * K + k0 + kb);
    v8h a0lo = a0p[0], a0hi = a0p[2];
    v8h a1lo = a1p[0], a1hi = a1p[2];
    v16h a0, a1;
    #pragma unroll
    for (int e = 0; e < 8; ++e) {
      a0[e] = a0lo[e]; a0[e + 8] = a0hi[e];
      a1[e] = a1lo[e]; a1[e + 8] = a1hi[e];
    }

    // 4 N-subtiles; each B fragment feeds both m-subtiles
    #pragma unroll
    for (int s = 0; s < 4; ++s) {
      const v8h* bp = (const v8h*)&Bt[p][(s * 16 + (lane & 15)) * BST + kb];
      v8h blo = bp[0];
      v8h bhi = bp[2];
      v16h bf;
      #pragma unroll
      for (int e = 0; e < 8; ++e) { bf[e] = blo[e]; bf[e + 8] = bhi[e]; }
      acc[0][s] = __builtin_amdgcn_wmma_f32_16x16x32_f16(
          false, a0, false, bf, (short)0, acc[0][s], false, false);
      acc[1][s] = __builtin_amdgcn_wmma_f32_16x16x32_f16(
          false, a1, false, bf, (short)0, acc[1][s], false, false);
    }

    if (kn < K) {
      *(v8h*)&Bt[p ^ 1][fn * BST + fko] = nxt;
      __builtin_prefetch(Wm + (size_t)mRow * K + kn + kb, 0, 3);
    }
    __syncthreads();
    p ^= 1;
  }

  // C/D layout: lanes 0-15 -> M=r, lanes 16-31 -> M=r+8; N = lane&15
  #pragma unroll
  for (int ms = 0; ms < 2; ++ms) {
    #pragma unroll
    for (int s = 0; s < 4; ++s) {
      const int nCol = nBase + s * 16 + (lane & 15);
      const int ob   = nCol / NP;
      const int os   = nCol - ob * NP;
      #pragma unroll
      for (int r = 0; r < 8; ++r) {
        int oc = ms * 128 + wave * 16 + r + ((lane >> 4) * 8);
        float v = acc[ms][s][r] + bias[oc];
        if (MODE == 0) {
          v = fmaxf(v, 0.f);
          OutH[((size_t)ob * 256 + oc) * NP + os] = (_Float16)v;
        } else {
          OutF[((size_t)ob * 256 + oc) * NP + os] = v;
        }
      }
    }
  }
}

// ------------------------- squash primary capsules -------------------------
// pc (B,256,36) -> u (B,1152,8) with the reshape/permute folded in.
__global__ __launch_bounds__(256)
void k_squash_u(const float* __restrict__ pc, float* __restrict__ u)
{
  int idx = blockIdx.x * 256 + threadIdx.x;              // (b,i)
  if (idx >= 512 * 1152) return;
  int i = idx % 1152, b = idx / 1152;
  int cap = i & 31, s = i >> 5;                          // i = s*32 + cap
  float t[8];
  float sq = 0.f;
  #pragma unroll
  for (int k = 0; k < 8; ++k) {
    t[k] = pc[((size_t)b * 256 + cap * 8 + k) * 36 + s];
    sq += t[k] * t[k];
  }
  float sc = (sq / (1.f + sq)) * rsqrtf(sq + 1e-8f);
  #pragma unroll
  for (int k = 0; k < 8; ++k) u[(size_t)idx * 8 + k] = t[k] * sc;
}

// ------------------------------- routing -----------------------------------
__global__ __launch_bounds__(256)
void k_zero(float* __restrict__ p, int n)
{
  int i = blockIdx.x * 256 + threadIdx.x;
  if (i < n) p[i] = 0.f;
}

__global__ __launch_bounds__(256)
void k_softmax_c(const float* __restrict__ bij, float* __restrict__ cij)
{
  int idx = blockIdx.x * 256 + threadIdx.x;              // (b,i)
  if (idx >= 512 * 1152) return;
  const float* p = bij + (size_t)idx * 10;
  float m = p[0];
  #pragma unroll
  for (int j = 1; j < 10; ++j) m = fmaxf(m, p[j]);
  float e[10], sum = 0.f;
  #pragma unroll
  for (int j = 0; j < 10; ++j) { e[j] = __expf(p[j] - m); sum += e[j]; }
  float inv = 1.f / sum;
  float* q = cij + (size_t)idx * 10;
  #pragma unroll
  for (int j = 0; j < 10; ++j) q[j] = e[j] * inv;
}

// s_j[b,j,d] = sum_i c[b,i,j] * (W[i,j,d,:] . u[b,i,:])  (u_hat recomputed)
__global__ __launch_bounds__(256)
void k_route_sj(const float* __restrict__ Wd, const float* __restrict__ u,
                const float* __restrict__ cij, float* __restrict__ sj)
{
  int b = blockIdx.x / 9, chunk = blockIdx.x % 9;
  int i0 = chunk * 128;
  int t = threadIdx.x;
  if (t >= 160) return;
  int j = t >> 4, d = t & 15;
  float acc = 0.f;
  for (int ii = 0; ii < 128; ++ii) {
    int i = i0 + ii;
    const float* wp = Wd + (((size_t)i * 10 + j) * 16 + d) * 8;
    const float* up = u + ((size_t)b * 1152 + i) * 8;
    float uh = 0.f;
    #pragma unroll
    for (int k = 0; k < 8; ++k) uh += wp[k] * up[k];
    acc += cij[((size_t)b * 1152 + i) * 10 + j] * uh;
  }
  atomicAdd(&sj[((size_t)b * 10 + j) * 16 + d], acc);
}

__global__ __launch_bounds__(256)
void k_squash_v(const float* __restrict__ sj, float* __restrict__ vj)
{
  int idx = blockIdx.x * 256 + threadIdx.x;              // (b,j)
  if (idx >= 512 * 10) return;
  const float* p = sj + (size_t)idx * 16;
  float sq = 0.f;
  #pragma unroll
  for (int d = 0; d < 16; ++d) sq += p[d] * p[d];
  float sc = (sq / (1.f + sq)) * rsqrtf(sq + 1e-8f);
  float* q = vj + (size_t)idx * 16;
  #pragma unroll
  for (int d = 0; d < 16; ++d) q[d] = p[d] * sc;
}

// b_ij[b,i,j] += sum_d u_hat[b,i,j,d] * v[b,j,d]  (u_hat recomputed)
__global__ __launch_bounds__(256)
void k_route_bij(const float* __restrict__ Wd, const float* __restrict__ u,
                 const float* __restrict__ vj, float* __restrict__ bij)
{
  int idx = blockIdx.x * 256 + threadIdx.x;              // (b,i,j)
  if (idx >= 512 * 1152 * 10) return;
  int j = idx % 10;
  int t = idx / 10;
  int i = t % 1152;
  int b = t / 1152;
  const float* up = u + ((size_t)b * 1152 + i) * 8;
  const float* vp = vj + ((size_t)b * 10 + j) * 16;
  const float* wp = Wd + ((size_t)i * 10 + j) * 128;
  float uu[8];
  #pragma unroll
  for (int k = 0; k < 8; ++k) uu[k] = up[k];
  float acc = 0.f;
  #pragma unroll
  for (int d = 0; d < 16; ++d) {
    float uh = 0.f;
    #pragma unroll
    for (int k = 0; k < 8; ++k) uh += wp[d * 8 + k] * uu[k];
    acc += uh * vp[d];
  }
  bij[idx] += acc;
}

__global__ __launch_bounds__(256)
void k_norm(const float* __restrict__ vj, float* __restrict__ out)
{
  int idx = blockIdx.x * 256 + threadIdx.x;              // (b,j)
  if (idx >= 512 * 10) return;
  const float* p = vj + (size_t)idx * 16;
  float sq = 0.f;
  #pragma unroll
  for (int d = 0; d < 16; ++d) sq += p[d] * p[d];
  out[idx] = sqrtf(sq);
}

// ------------------------------- launcher ----------------------------------
extern "C" void kernel_launch(void* const* d_in, const int* in_sizes, int n_in,
                              void* d_out, int out_size, void* d_ws, size_t ws_size,
                              hipStream_t stream)
{
  const float* x   = (const float*)d_in[0];
  const float* c1w = (const float*)d_in[1];
  const float* c1b = (const float*)d_in[2];
  const float* bng = (const float*)d_in[3];
  const float* bnb = (const float*)d_in[4];
  const float* bnm = (const float*)d_in[5];
  const float* bnv = (const float*)d_in[6];
  const float* c2w = (const float*)d_in[7];
  const float* c2b = (const float*)d_in[8];
  const float* pw  = (const float*)d_in[9];
  const float* pb  = (const float*)d_in[10];
  const float* Wd  = (const float*)d_in[11];
  float* out = (float*)d_out;

  char* ws = (char*)d_ws;
  size_t off = 0;
  auto carve = [&](size_t bytes) -> char* {
    char* p = ws + off;
    off += (bytes + 255) & ~(size_t)255;
    return p;
  };
  _Float16* h1  = (_Float16*)carve(512ull * 128 * 784 * 2);   // conv1 out (f16)
  _Float16* h2  = (_Float16*)carve(512ull * 256 * 400 * 2);   // conv2 out (f16)
  _Float16* w2h = (_Float16*)carve(256ull * 10368 * 2);       // conv2 w  (f16, K-permuted)
  _Float16* wph = (_Float16*)carve(256ull * 20736 * 2);       // pcaps w  (f16, K-permuted)
  float* pc  = (float*)carve(512ull * 256 * 36 * 4);          // pcaps out
  float* u   = (float*)carve(512ull * 1152 * 8 * 4);          // squashed caps
  float* bij = (float*)carve(512ull * 1152 * 10 * 4);
  float* cij = (float*)carve(512ull * 1152 * 10 * 4);
  float* sj  = (float*)carve(512ull * 160 * 4);
  float* vj  = (float*)carve(512ull * 160 * 4);

  // weight repack (oc,ic,kk) -> (oc,kk,ic) + f32->f16
  k_repack_w<<<(256 * 10368 + 255) / 256, 256, 0, stream>>>(c2w, w2h, 128);
  k_repack_w<<<(256 * 20736 + 255) / 256, 256, 0, stream>>>(pw, wph, 256);

  // conv1 + BN + ReLU
  k_conv1<<<(512 * 128 * 784) / 256, 256, 0, stream>>>(x, c1w, c1b, bng, bnb, bnm, bnv, h1);

  // conv2 (WMMA implicit GEMM): N tiles = 512*400/64, full M in block
  k_convgemm<0><<<512 * 400 / 64, 256, 0, stream>>>(w2h, h1, c2b, h2, nullptr);

  // primary caps (WMMA implicit GEMM): N tiles = 512*36/64
  k_convgemm<1><<<512 * 36 / 64, 256, 0, stream>>>(wph, h2, pb, nullptr, pc);

  // squash -> u (B,1152,8)
  k_squash_u<<<(512 * 1152 + 255) / 256, 256, 0, stream>>>(pc, u);

  // routing (3 iterations; last b_ij update is dead in the reference)
  k_zero<<<(512 * 1152 * 10 + 255) / 256, 256, 0, stream>>>(bij, 512 * 1152 * 10);
  for (int it = 0; it < 3; ++it) {
    k_softmax_c<<<(512 * 1152 + 255) / 256, 256, 0, stream>>>(bij, cij);
    k_zero<<<(512 * 160 + 255) / 256, 256, 0, stream>>>(sj, 512 * 160);
    k_route_sj<<<512 * 9, 256, 0, stream>>>(Wd, u, cij, sj);
    k_squash_v<<<(5120 + 255) / 256, 256, 0, stream>>>(sj, vj);
    if (it < 2)
      k_route_bij<<<(512 * 1152 * 10 + 255) / 256, 256, 0, stream>>>(Wd, u, vj, bij);
  }

  // ||v|| -> (B,10)
  k_norm<<<(5120 + 255) / 256, 256, 0, stream>>>(vj, out);
}
